// Head_74337293959213
// MI455X (gfx1250) — matrix operand order repaired and checked by reading, not compiled, not run on hardware
//
#include <hip/hip_runtime.h>
#include <math.h>

typedef __attribute__((ext_vector_type(16))) __bf16 v16bf;
typedef __attribute__((ext_vector_type(8)))  __bf16 v8bf;   // 16 bytes
typedef __attribute__((ext_vector_type(8)))  float  v8f;

#define B_  4
#define T_  4096
#define C_  768
#define H_  64
#define M_  (B_ * T_)   // 16384 rows

__device__ __forceinline__ v16bf cat16(v8bf lo, v8bf hi) {
    return __builtin_shufflevector(lo, hi, 0,1,2,3,4,5,6,7,8,9,10,11,12,13,14,15);
}
__device__ __forceinline__ v8bf ldsv8(const __bf16* p) {
    return *reinterpret_cast<const v8bf*>(p);
}

// ws layout (bf16):
//   qkv Q : [M_][H_]          at 0
//   qkv K : [M_][H_]          at M_*H_
//   Vt_g  : [B_][H_][T_]      at 2*M_*H_   (V transposed)
//   Wt_g  : [3][H_][C_]       at 3*M_*H_   (W transposed, bf16)

// ---------------------------------------------------------------------------
// Kernel 0: one-time W convert+transpose  Wt_g[which][h][k] = W[k][h]
// ---------------------------------------------------------------------------
__global__ __launch_bounds__(256) void prep_w(
    const float* __restrict__ Wq,
    const float* __restrict__ Wk,
    const float* __restrict__ Wv,
    __bf16* __restrict__ wt)
{
    int idx = blockIdx.x * 256 + threadIdx.x;       // over 3*H_*C_
    if (idx >= 3 * H_ * C_) return;
    int which = idx / (H_ * C_);
    int rem   = idx % (H_ * C_);
    int h = rem / C_, k = rem % C_;
    const float* W = (which == 0) ? Wq : ((which == 1) ? Wk : Wv);
    wt[idx] = (__bf16)W[(size_t)k * H_ + h];
}

// ---------------------------------------------------------------------------
// Kernel 1: Q/K/V projection (double-buffered).  256 threads = 8 waves;
// each wave owns 16 rows x 64 cols.  Q pre-scaled by H^-0.5; V stored
// transposed ([B][H][T]) for the attention kernel's B-fragments.
// ---------------------------------------------------------------------------
__global__ __launch_bounds__(256) void proj_kernel(
    const float* __restrict__ x,
    __bf16* __restrict__ qkv,       // Q then K
    __bf16* __restrict__ vt,        // [B][H][T]
    const __bf16* __restrict__ wt)  // [3][H][C]
{
    const int which = blockIdx.y;
    const __bf16* wbase = wt + (size_t)which * H_ * C_;

    // pitch 40 bf16 = 80B = 20 banks: conflict-free b128 fragment reads
    __shared__ __align__(16) __bf16 As[2][128][40];  // x tile, bf16
    __shared__ __align__(16) __bf16 Wt[2][64][40];   // W transposed [h][k]

    const int tid  = threadIdx.x;
    const int lane = tid & 31;
    const int w    = tid >> 5;
    const int lg   = lane >> 4;
    const int n    = lane & 15;

    const int rowBase = blockIdx.x * 128;
    const int xr = tid >> 1, xc0 = (tid & 1) * 16;
    const int wh = tid >> 2, wq  = (tid & 3) * 8;
    const float*  xrow = x + (size_t)(rowBase + xr) * C_ + xc0;
    const __bf16* wrow = wbase + (size_t)wh * C_ + wq;
    const int NKC = C_ / 32;

    float xreg[16];
    v8bf  wreg;
    v8f   acc[4] = {};

    // prologue: fetch + stage chunk 0
    #pragma unroll
    for (int j = 0; j < 16; ++j) xreg[j] = xrow[j];
    wreg = *reinterpret_cast<const v8bf*>(wrow);
    {
        v8bf t0, t1;
        #pragma unroll
        for (int j = 0; j < 8; ++j) { t0[j] = (__bf16)xreg[j]; t1[j] = (__bf16)xreg[8 + j]; }
        *reinterpret_cast<v8bf*>(&As[0][xr][xc0])     = t0;
        *reinterpret_cast<v8bf*>(&As[0][xr][xc0 + 8]) = t1;
        *reinterpret_cast<v8bf*>(&Wt[0][wh][wq])      = wreg;
    }
    __syncthreads();

    for (int kc = 0; kc < NKC; ++kc) {
        const int cur = kc & 1;
        // prefetch next chunk into registers (overlaps with WMMAs below)
        if (kc + 1 < NKC) {
            #pragma unroll
            for (int j = 0; j < 16; ++j) xreg[j] = xrow[(kc + 1) * 32 + j];
            wreg = *reinterpret_cast<const v8bf*>(wrow + (kc + 1) * 32);
        }

        // compute on current buffer: batch all fragment loads, then 4 WMMAs
        v16bf a = cat16(ldsv8(&As[cur][w * 16 + n][lg * 16]),
                        ldsv8(&As[cur][w * 16 + n][lg * 16 + 8]));
        v16bf bf0 = cat16(ldsv8(&Wt[cur][ 0 + n][lg * 16]), ldsv8(&Wt[cur][ 0 + n][lg * 16 + 8]));
        v16bf bf1 = cat16(ldsv8(&Wt[cur][16 + n][lg * 16]), ldsv8(&Wt[cur][16 + n][lg * 16 + 8]));
        v16bf bf2 = cat16(ldsv8(&Wt[cur][32 + n][lg * 16]), ldsv8(&Wt[cur][32 + n][lg * 16 + 8]));
        v16bf bf3 = cat16(ldsv8(&Wt[cur][48 + n][lg * 16]), ldsv8(&Wt[cur][48 + n][lg * 16 + 8]));
        acc[0] = __builtin_amdgcn_wmma_f32_16x16x32_bf16(false, a, false, bf0, (short)0, acc[0], false, false);
        acc[1] = __builtin_amdgcn_wmma_f32_16x16x32_bf16(false, a, false, bf1, (short)0, acc[1], false, false);
        acc[2] = __builtin_amdgcn_wmma_f32_16x16x32_bf16(false, a, false, bf2, (short)0, acc[2], false, false);
        acc[3] = __builtin_amdgcn_wmma_f32_16x16x32_bf16(false, a, false, bf3, (short)0, acc[3], false, false);

        // stage next chunk into the other buffer
        if (kc + 1 < NKC) {
            const int nxt = cur ^ 1;
            v8bf t0, t1;
            #pragma unroll
            for (int j = 0; j < 8; ++j) { t0[j] = (__bf16)xreg[j]; t1[j] = (__bf16)xreg[8 + j]; }
            *reinterpret_cast<v8bf*>(&As[nxt][xr][xc0])     = t0;
            *reinterpret_cast<v8bf*>(&As[nxt][xr][xc0 + 8]) = t1;
            *reinterpret_cast<v8bf*>(&Wt[nxt][wh][wq])      = wreg;
        }
        __syncthreads();
    }

    // C layout: VGPR r -> row r + 8*lg, col = 16*f + n
    if (which < 2) {
        __bf16* out = qkv + (size_t)which * M_ * H_;
        const float scale = (which == 0) ? 0.125f : 1.0f;   // fold H^-0.5 into Q
        #pragma unroll
        for (int f = 0; f < 4; ++f)
            #pragma unroll
            for (int r = 0; r < 8; ++r) {
                int row = rowBase + w * 16 + r + lg * 8;
                out[(size_t)row * H_ + f * 16 + n] = (__bf16)(acc[f][r] * scale);
            }
    } else {
        // V stored transposed: vt[(batch*H + h)*T + t]
        #pragma unroll
        for (int f = 0; f < 4; ++f)
            #pragma unroll
            for (int r = 0; r < 8; ++r) {
                int row = rowBase + w * 16 + r + lg * 8;
                int batch = row >> 12, t = row & (T_ - 1);
                vt[((size_t)batch * H_ + f * 16 + n) * T_ + t] = (__bf16)acc[f][r];
            }
    }
}

// ---------------------------------------------------------------------------
// Kernel 2: causal flash attention (double-buffered K/V).
// 128 queries / workgroup, 8 waves x 16 rows.
// ---------------------------------------------------------------------------
__global__ __launch_bounds__(256) void attn_kernel(
    const __bf16* __restrict__ qkv,
    const __bf16* __restrict__ vt,    // [B][H][T]
    float* __restrict__ out)
{
    const __bf16* Qb = qkv;
    const __bf16* Kb = qkv + (size_t)1 * M_ * H_;

    const int b  = blockIdx.y;
    const int ts = blockIdx.x * 128;    // first query of tile

    __shared__ __align__(16) __bf16 Qs[128][64];
    __shared__ __align__(16) __bf16 Ks[2][32][64];     // key-major
    __shared__ __align__(16) __bf16 Vt[2][64][40];     // [h][key-in-chunk]
    __shared__ __align__(16) __bf16 Ps[8][16][40];     // per-wave P bounce

    const int tid  = threadIdx.x;
    const int lane = tid & 31;
    const int w    = tid >> 5;
    const int lg   = lane >> 4;
    const int n    = lane & 15;

    // stage Q tile (bf16, pre-scaled) with 16B vector copies
    {
        int r  = tid >> 1;
        int c0 = (tid & 1) * 32;
        const __bf16* src = Qb + ((size_t)b * T_ + ts + r) * H_ + c0;
        #pragma unroll
        for (int j = 0; j < 4; ++j)
            *reinterpret_cast<v8bf*>(&Qs[r][c0 + j * 8]) =
                *reinterpret_cast<const v8bf*>(src + j * 8);
    }
    __syncthreads();

    // Q as two A-fragments (K = 0..31 and 32..63), kept in registers
    const __bf16* qrow = &Qs[w * 16 + n][0];
    v16bf qa0 = cat16(ldsv8(qrow + lg * 16),      ldsv8(qrow + lg * 16 + 8));
    v16bf qa1 = cat16(ldsv8(qrow + 32 + lg * 16), ldsv8(qrow + 32 + lg * 16 + 8));

    float mi[8], li[8];
    #pragma unroll
    for (int r = 0; r < 8; ++r) { mi[r] = -INFINITY; li[r] = 0.0f; }
    v8f acc[4] = {};

    const int nchunks = ts / 32 + 4;    // keys 0 .. ts+127

    // staging coordinates: K row-major, V from transposed global layout
    const int sr = tid >> 3;            // key 0..31
    const int sc = (tid & 7) * 8;       // h group for K
    const int vh = tid >> 2;            // h 0..63
    const int vq = (tid & 3) * 8;       // key group for V
    const __bf16* kbase = Kb + ((size_t)b * T_ + sr) * H_ + sc;
    const __bf16* vbase = vt + ((size_t)b * H_ + vh) * T_ + vq;

    v8bf kreg, vreg;
    // prologue: fetch + stage chunk 0 into buffer 0
    kreg = *reinterpret_cast<const v8bf*>(kbase);
    vreg = *reinterpret_cast<const v8bf*>(vbase);
    *reinterpret_cast<v8bf*>(&Ks[0][sr][sc]) = kreg;
    *reinterpret_cast<v8bf*>(&Vt[0][vh][vq]) = vreg;
    __syncthreads();

    for (int c = 0; c < nchunks; ++c) {
        const int cur = c & 1;
        const int kb  = c * 32;

        // prefetch next K/V chunk into registers (overlaps with math below)
        if (c + 1 < nchunks) {
            kreg = *reinterpret_cast<const v8bf*>(kbase + (size_t)(c + 1) * 32 * H_);
            vreg = *reinterpret_cast<const v8bf*>(vbase + (c + 1) * 32);
        }

        // S = Q K^T : batch all B-fragment loads, then 4 WMMAs
        v16bf bk00 = cat16(ldsv8(&Ks[cur][n][lg * 16]),           ldsv8(&Ks[cur][n][lg * 16 + 8]));
        v16bf bk01 = cat16(ldsv8(&Ks[cur][n][32 + lg * 16]),      ldsv8(&Ks[cur][n][32 + lg * 16 + 8]));
        v16bf bk10 = cat16(ldsv8(&Ks[cur][16 + n][lg * 16]),      ldsv8(&Ks[cur][16 + n][lg * 16 + 8]));
        v16bf bk11 = cat16(ldsv8(&Ks[cur][16 + n][32 + lg * 16]), ldsv8(&Ks[cur][16 + n][32 + lg * 16 + 8]));
        v8f s0 = {}, s1 = {};
        s0 = __builtin_amdgcn_wmma_f32_16x16x32_bf16(false, qa0, false, bk00, (short)0, s0, false, false);
        s0 = __builtin_amdgcn_wmma_f32_16x16x32_bf16(false, qa1, false, bk01, (short)0, s0, false, false);
        s1 = __builtin_amdgcn_wmma_f32_16x16x32_bf16(false, qa0, false, bk10, (short)0, s1, false, false);
        s1 = __builtin_amdgcn_wmma_f32_16x16x32_bf16(false, qa1, false, bk11, (short)0, s1, false, false);

        // causal mask + online softmax (each row lives in one 16-lane half)
        #pragma unroll
        for (int r = 0; r < 8; ++r) {
            int qg  = ts + w * 16 + r + lg * 8;
            float v0 = ((kb + n)      <= qg) ? s0[r] : -INFINITY;
            float v1 = ((kb + 16 + n) <= qg) ? s1[r] : -INFINITY;
            float mx = fmaxf(v0, v1);
            mx = fmaxf(mx, __shfl_xor(mx, 1));
            mx = fmaxf(mx, __shfl_xor(mx, 2));
            mx = fmaxf(mx, __shfl_xor(mx, 4));
            mx = fmaxf(mx, __shfl_xor(mx, 8));
            float mnew  = fmaxf(mi[r], mx);
            float alpha = __expf(mi[r] - mnew);
            float p0 = __expf(v0 - mnew);
            float p1 = __expf(v1 - mnew);
            float ls = p0 + p1;
            ls += __shfl_xor(ls, 1);
            ls += __shfl_xor(ls, 2);
            ls += __shfl_xor(ls, 4);
            ls += __shfl_xor(ls, 8);
            li[r] = li[r] * alpha + ls;
            mi[r] = mnew;
            acc[0][r] *= alpha; acc[1][r] *= alpha;
            acc[2][r] *= alpha; acc[3][r] *= alpha;
            Ps[w][r + lg * 8][n]      = (__bf16)p0;
            Ps[w][r + lg * 8][16 + n] = (__bf16)p1;
        }
        // wave-local LDS RAW: P stores must land before A-fragment reads
        asm volatile("s_wait_dscnt 0" ::: "memory");

        // O += P V : batch loads, then 4 WMMAs
        v16bf pa  = cat16(ldsv8(&Ps[w][n][lg * 16]), ldsv8(&Ps[w][n][lg * 16 + 8]));
        v16bf bv0 = cat16(ldsv8(&Vt[cur][ 0 + n][lg * 16]), ldsv8(&Vt[cur][ 0 + n][lg * 16 + 8]));
        v16bf bv1 = cat16(ldsv8(&Vt[cur][16 + n][lg * 16]), ldsv8(&Vt[cur][16 + n][lg * 16 + 8]));
        v16bf bv2 = cat16(ldsv8(&Vt[cur][32 + n][lg * 16]), ldsv8(&Vt[cur][32 + n][lg * 16 + 8]));
        v16bf bv3 = cat16(ldsv8(&Vt[cur][48 + n][lg * 16]), ldsv8(&Vt[cur][48 + n][lg * 16 + 8]));
        acc[0] = __builtin_amdgcn_wmma_f32_16x16x32_bf16(false, pa, false, bv0, (short)0, acc[0], false, false);
        acc[1] = __builtin_amdgcn_wmma_f32_16x16x32_bf16(false, pa, false, bv1, (short)0, acc[1], false, false);
        acc[2] = __builtin_amdgcn_wmma_f32_16x16x32_bf16(false, pa, false, bv2, (short)0, acc[2], false, false);
        acc[3] = __builtin_amdgcn_wmma_f32_16x16x32_bf16(false, pa, false, bv3, (short)0, acc[3], false, false);

        // stage next chunk into the other buffer, single barrier per iteration
        if (c + 1 < nchunks) {
            const int nxt = cur ^ 1;
            *reinterpret_cast<v8bf*>(&Ks[nxt][sr][sc]) = kreg;
            *reinterpret_cast<v8bf*>(&Vt[nxt][vh][vq]) = vreg;
        }
        __syncthreads();
    }

    // normalize and store fp32 output
    #pragma unroll
    for (int f = 0; f < 4; ++f)
        #pragma unroll
        for (int r = 0; r < 8; ++r) {
            int row = ts + w * 16 + r + lg * 8;
            out[((size_t)b * T_ + row) * H_ + f * 16 + n] = acc[f][r] / li[r];
        }
}

// ---------------------------------------------------------------------------
extern "C" void kernel_launch(void* const* d_in, const int* in_sizes, int n_in,
                              void* d_out, int out_size, void* d_ws, size_t ws_size,
                              hipStream_t stream)
{
    const float* x  = (const float*)d_in[0];
    const float* Wq = (const float*)d_in[1];
    const float* Wk = (const float*)d_in[2];
    const float* Wv = (const float*)d_in[3];

    __bf16* qkv = (__bf16*)d_ws;                    // Q,K: 2*[M_][H_]
    __bf16* vt  = qkv + (size_t)2 * M_ * H_;        // [B_][H_][T_]
    __bf16* wt  = qkv + (size_t)3 * M_ * H_;        // [3][H_][C_]

    prep_w<<<(3 * H_ * C_ + 255) / 256, 256, 0, stream>>>(Wq, Wk, Wv, wt);
    proj_kernel<<<dim3(M_ / 128, 3), 256, 0, stream>>>(x, qkv, vt, wt);
    attn_kernel<<<dim3(T_ / 128, B_), 256, 0, stream>>>(qkv, vt, (float*)d_out);
}
